// GNN_6949257085648
// MI455X (gfx1250) — matrix operand order repaired and checked by reading, not compiled, not run on hardware
//
#include <hip/hip_runtime.h>

#define NNODES 100000
#define NEDGES 1600000

typedef __attribute__((ext_vector_type(16))) _Float16 v16h;
typedef __attribute__((ext_vector_type(8)))  float    v8f;

// Guaranteed-native L2 float atomic add (non-returning -> STOREcnt only).
static __device__ __forceinline__ void atomic_add_f32(float* p, float v) {
    asm volatile("global_atomic_add_f32 %0, %1, off scope:SCOPE_DEV"
                 :: "v"(p), "v"(v) : "memory");
}

// ---------------- zero workspace ----------------
__global__ void zero_kernel(float4* __restrict__ p, long long n4) {
    long long i = (long long)blockIdx.x * blockDim.x + threadIdx.x;
    if (i < n4) p[i] = make_float4(0.f, 0.f, 0.f, 0.f);
}

// ---------------- weight convert: transpose + f16 + WMMA-fragment swizzle ----------------
// Packed layout: Bp[((ntile*4 + kk)*32 + lane)*16 + j], j = fragment half index.
// Inverse map: grp=lane>>4, l16=lane&15, n = ntile*16+l16, k = kk*32 + 16*grp + j  (j=2v+sub).
__global__ void convw_kernel(const float* __restrict__ W1l, const float* __restrict__ W1r,
                             const float* __restrict__ W2l, const float* __restrict__ W2r,
                             _Float16* __restrict__ B1p, _Float16* __restrict__ B2lp,
                             _Float16* __restrict__ B2rp) {
    int t = blockIdx.x * blockDim.x + threadIdx.x;
    int stride = gridDim.x * blockDim.x;
    // B1p: 8 ntiles x 4 kk x 32 lanes x 16 = 16384 halves ; B[k][n]: k<64->W1l[n][k], else W1r
    for (int i = t; i < 8 * 4 * 32 * 16; i += stride) {
        int j = i & 15, lane = (i >> 4) & 31, kk = (i >> 9) & 3, ntile = i >> 11;
        int grp = lane >> 4, l16 = lane & 15;
        int n = ntile * 16 + l16;
        int k = kk * 32 + 16 * grp + j;
        float v = (k < 64) ? W1l[n * 64 + k] : W1r[n * 64 + (k - 64)];
        B1p[i] = (_Float16)v;
    }
    // B2: 4 ntiles x 4 kk x 32 x 16 = 8192 halves each ; B[k][n] = W[n*128+k]
    for (int i = t; i < 4 * 4 * 32 * 16; i += stride) {
        int j = i & 15, lane = (i >> 4) & 31, kk = (i >> 9) & 3, ntile = i >> 11;
        int grp = lane >> 4, l16 = lane & 15;
        int n = ntile * 16 + l16;
        int k = kk * 32 + 16 * grp + j;
        B2lp[i] = (_Float16)W2l[n * 128 + k];
        B2rp[i] = (_Float16)W2r[n * 128 + k];
    }
}

// ---------------- edge scatter: 1 wave (32 lanes) per edge, 64 f32 dims ----------------
template <bool WITH_DEG>
__global__ void scatter_kernel(const float* __restrict__ feat,   // [N,64]
                               const int* __restrict__ eidx,     // [2,E]
                               float* __restrict__ sums,         // [N,64]
                               float* __restrict__ deg) {
    long long g = (long long)blockIdx.x * blockDim.x + threadIdx.x;
    long long e = g >> 5;
    int lane = (int)(g & 31);
    if (e >= NEDGES) return;
    int src = eidx[e];
    int dst = eidx[NEDGES + e];
    float v0 = feat[src * 64 + lane];
    float v1 = feat[src * 64 + lane + 32];
    atomic_add_f32(&sums[dst * 64 + lane], v0);
    atomic_add_f32(&sums[dst * 64 + lane + 32], v1);
    if constexpr (WITH_DEG) {
        if (lane == 0) atomic_add_f32(&deg[dst], 1.0f);
    }
}

// ---- A fragment from LDS (16x32 f16, contiguous k = k0+16*grp+s, same order as packed B) ----
static __device__ __forceinline__ v16h load_a_frag(const _Float16* row, int k0, int grp) {
    v16h a;
#pragma unroll
    for (int v = 0; v < 8; ++v) {
        int ka = 2 * v + 16 * grp + k0;
        a[2 * v]     = row[ka];
        a[2 * v + 1] = row[ka + 1];
    }
    return a;
}
// ---- B fragment: pre-swizzled, one aligned 32B vector load ----
static __device__ __forceinline__ v16h load_b_packed(const _Float16* __restrict__ Bp,
                                                     int ntile, int kk, int lane) {
    const v16h* p = reinterpret_cast<const v16h*>(Bp) + ((ntile * 4 + kk) * 32 + lane);
    return *p;
}

// ---------------- layer-1 GEMM: h = relu([mean|x] @ B1 + b1), 16 rows/block ----------------
__global__ __launch_bounds__(256) void gemm1_kernel(const float* __restrict__ sum1,
                                                    const float* __restrict__ deg,
                                                    const float* __restrict__ x,
                                                    const _Float16* __restrict__ B1p,
                                                    const float* __restrict__ b1,
                                                    float* __restrict__ h) {
    __shared__ _Float16 As[16][136];
    __shared__ float invd[16];
    const int tid = threadIdx.x;
    const int row0 = blockIdx.x * 16;

    if (tid < 16) invd[tid] = 1.0f / fmaxf(deg[row0 + tid], 1.0f);
    __syncthreads();

#pragma unroll
    for (int i = 0; i < 2; ++i) {
        int c = i * 256 + tid;          // 512 float4 chunks: 16 rows x 32 chunks
        int r = c >> 5;
        int kc = (c & 31) * 4;
        float4 v;
        if (kc < 64) {
            v = *(const float4*)&sum1[(row0 + r) * 64 + kc];
            float s = invd[r];
            v.x *= s; v.y *= s; v.z *= s; v.w *= s;
        } else {
            v = *(const float4*)&x[(row0 + r) * 64 + (kc - 64)];
        }
        As[r][kc + 0] = (_Float16)v.x; As[r][kc + 1] = (_Float16)v.y;
        As[r][kc + 2] = (_Float16)v.z; As[r][kc + 3] = (_Float16)v.w;
    }
    __syncthreads();

    const int wave = tid >> 5;           // 8 waves -> 8 N-tiles of 16 (128 cols)
    const int lane = tid & 31;
    const int grp = lane >> 4, l16 = lane & 15;

    v8f acc = {};
#pragma unroll
    for (int kk = 0; kk < 4; ++kk) {
        v16h a = load_a_frag(&As[l16][0], kk * 32, grp);
        v16h b = load_b_packed(B1p, wave, kk, lane);
        acc = __builtin_amdgcn_wmma_f32_16x16x32_f16(false, a, false, b, (short)0, acc,
                                                     false, false);
    }
    const int ncol = wave * 16 + l16;
    const float bias = b1[ncol];
#pragma unroll
    for (int v = 0; v < 8; ++v) {
        int m = v + 8 * grp;
        h[(row0 + m) * 128 + ncol] = fmaxf(acc[v] + bias, 0.0f);
    }
}

// ---------------- layer-2 GEMMs: [N,128] @ [128,64]; 32 rows/block, 8 waves ----------------
// FINAL=false: out = h @ B2l                       (pre-aggregation projection g)
// FINAL=true : out = relu(h @ B2r + sum2/deg + b2) (final output)
template <bool FINAL>
__global__ __launch_bounds__(256) void gemm2_kernel(const float* __restrict__ hin, // [N,128]
                                                    const _Float16* __restrict__ Bp,
                                                    const float* __restrict__ sum2,
                                                    const float* __restrict__ deg,
                                                    const float* __restrict__ b2,
                                                    float* __restrict__ out) {   // [N,64]
    __shared__ _Float16 As[32][136];
    const int tid = threadIdx.x;
    const int row0 = blockIdx.x * 32;

#pragma unroll
    for (int i = 0; i < 4; ++i) {
        int c = i * 256 + tid;          // 1024 float4 chunks: 32 rows x 32 chunks
        int r = c >> 5;
        int kc = (c & 31) * 4;
        float4 v = *(const float4*)&hin[(row0 + r) * 128 + kc];
        As[r][kc + 0] = (_Float16)v.x; As[r][kc + 1] = (_Float16)v.y;
        As[r][kc + 2] = (_Float16)v.z; As[r][kc + 3] = (_Float16)v.w;
    }
    __syncthreads();

    const int wave = tid >> 5;
    const int sub = wave >> 2;           // which 16-row half
    const int ntile = wave & 3;          // 4 N-tiles cover 64 cols
    const int lane = tid & 31;
    const int grp = lane >> 4, l16 = lane & 15;

    v8f acc = {};
#pragma unroll
    for (int kk = 0; kk < 4; ++kk) {
        v16h a = load_a_frag(&As[sub * 16 + l16][0], kk * 32, grp);
        v16h b = load_b_packed(Bp, ntile, kk, lane);
        acc = __builtin_amdgcn_wmma_f32_16x16x32_f16(false, a, false, b, (short)0, acc,
                                                     false, false);
    }
    const int ncol = ntile * 16 + l16;
#pragma unroll
    for (int v = 0; v < 8; ++v) {
        int node = row0 + sub * 16 + v + 8 * grp;
        float val = acc[v];
        if constexpr (FINAL) {
            float d = fmaxf(deg[node], 1.0f);
            val += sum2[node * 64 + ncol] / d + b2[ncol];
            val = fmaxf(val, 0.0f);
        }
        out[node * 64 + ncol] = val;
    }
}

extern "C" void kernel_launch(void* const* d_in, const int* in_sizes, int n_in,
                              void* d_out, int out_size, void* d_ws, size_t ws_size,
                              hipStream_t stream) {
    (void)in_sizes; (void)n_in; (void)out_size; (void)ws_size;
    const float* x   = (const float*)d_in[0];
    const int*   ei  = (const int*)d_in[1];
    const float* W1l = (const float*)d_in[2];
    const float* b1  = (const float*)d_in[3];
    const float* W1r = (const float*)d_in[4];
    const float* W2l = (const float*)d_in[5];
    const float* b2  = (const float*)d_in[6];
    const float* W2r = (const float*)d_in[7];
    float* out = (float*)d_out;

    auto al = [](size_t v) { return (v + 255) & ~(size_t)255; };
    char* w = (char*)d_ws;
    size_t off = 0;
    float* deg  = (float*)(w + off); off += al((size_t)NNODES * 4);
    float* sum1 = (float*)(w + off); off += al((size_t)NNODES * 64 * 4);
    float* sum2 = (float*)(w + off); off += al((size_t)NNODES * 64 * 4);
    size_t zero_bytes = off;                       // deg+sum1+sum2 contiguous
    float* h = (float*)(w + off); off += al((size_t)NNODES * 128 * 4);
    float* g = (float*)(w + off); off += al((size_t)NNODES * 64 * 4);
    _Float16* B1p  = (_Float16*)(w + off); off += al((size_t)8 * 4 * 32 * 16 * 2);
    _Float16* B2lp = (_Float16*)(w + off); off += al((size_t)4 * 4 * 32 * 16 * 2);
    _Float16* B2rp = (_Float16*)(w + off); off += al((size_t)4 * 4 * 32 * 16 * 2);

    // 1) zero accumulators (re-zeroed every call: graph replays don't re-poison)
    long long n4 = (long long)(zero_bytes / 16);
    zero_kernel<<<(unsigned)((n4 + 255) / 256), 256, 0, stream>>>((float4*)d_ws, n4);

    // 2) f16 transposed + fragment-swizzled weights
    convw_kernel<<<64, 256, 0, stream>>>(W1l, W1r, W2l, W2r, B1p, B2lp, B2rp);

    // 3) layer-1 aggregation: sum1 += x[src] ; deg += 1
    long long sthreads = (long long)NEDGES * 32;
    scatter_kernel<true><<<(unsigned)(sthreads / 256), 256, 0, stream>>>(x, ei, sum1, deg);

    // 4) h = relu([sum1/deg | x] @ [W1_l^T; W1_r^T] + b1)
    gemm1_kernel<<<NNODES / 16, 256, 0, stream>>>(sum1, deg, x, B1p, b1, h);

    // 5) g = h @ W2_l^T   (project BEFORE aggregation: 64-dim scatter, not 128)
    gemm2_kernel<false><<<NNODES / 32, 256, 0, stream>>>(h, B2lp, nullptr, nullptr, nullptr, g);

    // 6) layer-2 aggregation on g
    scatter_kernel<false><<<(unsigned)(sthreads / 256), 256, 0, stream>>>(g, ei, sum2, nullptr);

    // 7) out = relu(sum2/deg + b2 + h @ W2_r^T)
    gemm2_kernel<true><<<NNODES / 32, 256, 0, stream>>>(h, B2rp, sum2, deg, b2, out);
}
